// DeepSeekMoE_86586540688037
// MI455X (gfx1250) — matrix operand order repaired and checked by reading, not compiled, not run on hardware
//
#include <hip/hip_runtime.h>
#include <hip/hip_bf16.h>

typedef __attribute__((ext_vector_type(8)))  float  v8f;
typedef __attribute__((ext_vector_type(16))) __bf16 v16bf;

#define TROWS 32   // tokens per expert-kernel block
#define D_DIM 512
#define H_DIM 64
#define O_DIM 512
#define E_NUM 16
#define T_TOK 8192

__device__ inline __bf16 f2bf(float f) {
    union { float f; unsigned u; } a; a.f = f;
    unsigned u = a.u;
    unsigned r = (u + 0x7FFFu + ((u >> 16) & 1u)) >> 16;
    union { unsigned short s; __bf16 b; } o; o.s = (unsigned short)r;
    return o.b;
}

// A fragment (16x32 bf16): lane<16 holds row M=lane, K = k0+{0..7, 16..23};
// lane>=16 holds row M=lane-16, K = k0+{8..15, 24..31}. LDS layout [row][k], stride S.
__device__ inline v16bf ld_fragA(const __bf16* p, int S, int m0, int k0, int lane) {
    int r  = m0 + (lane & 15);
    int kb = k0 + ((lane >> 4) << 3);
    const __bf16* q = p + r * S + kb;
    union { v16bf v; __bf16 e[16]; } u;
#pragma unroll
    for (int i = 0; i < 8; ++i) { u.e[i] = q[i]; u.e[i + 8] = q[i + 16]; }
    return u.v;
}

// B fragment (32x16 bf16): lane<16 holds col N=lane, K = k0+{0..7, 16..23};
// lane>=16 holds col N=lane-16, K = k0+{8..15, 24..31}. LDS stored transposed [n][k], stride S.
__device__ inline v16bf ld_fragB(const __bf16* p, int S, int n0, int k0, int lane) {
    int c  = n0 + (lane & 15);
    int kb = k0 + ((lane >> 4) << 3);
    const __bf16* q = p + c * S + kb;
    union { v16bf v; __bf16 e[16]; } u;
#pragma unroll
    for (int i = 0; i < 8; ++i) { u.e[i] = q[i]; u.e[i + 8] = q[i + 16]; }
    return u.v;
}

__device__ inline v8f wmma_bf16(v16bf a, v16bf b, v8f c) {
    return __builtin_amdgcn_wmma_f32_16x16x32_bf16(false, a, false, b, (short)0, c,
                                                   false, false);
}

// ---------------- Kernel 1: gating -> masked top-3 gate values ----------------
__global__ __launch_bounds__(64) void gate_kernel(
    const float* __restrict__ x,  const float* __restrict__ gW1,
    const float* __restrict__ gb1, const float* __restrict__ gW2,
    const float* __restrict__ gb2, float* __restrict__ gm)
{
    __shared__ float hsm[H_DIM];
    __shared__ float lg[E_NUM];
    const int t = blockIdx.x;
    const int tid = threadIdx.x;
    const float* xt = x + (size_t)t * D_DIM;

    float s = gb1[tid];
    for (int d = 0; d < D_DIM; ++d) s += xt[d] * gW1[d * H_DIM + tid];
    hsm[tid] = s > 0.f ? s : 0.f;
    __syncthreads();

    if (tid < E_NUM) {
        float l = gb2[tid];
        for (int j = 0; j < H_DIM; ++j) l += hsm[j] * gW2[j * E_NUM + tid];
        lg[tid] = l;
    }
    __syncthreads();

    if (tid == 0) {
        float mx = lg[0];
        for (int e = 1; e < E_NUM; ++e) mx = fmaxf(mx, lg[e]);
        float p[E_NUM]; float sum = 0.f;
        for (int e = 0; e < E_NUM; ++e) { p[e] = __expf(lg[e] - mx); sum += p[e]; }
        float inv = 1.f / sum;
        for (int e = 0; e < E_NUM; ++e) p[e] *= inv;
        float outg[E_NUM];
        for (int e = 0; e < E_NUM; ++e) outg[e] = 0.f;
        int u0 = -1, u1 = -1;
        for (int kk = 0; kk < 3; ++kk) {         // stable top-3 (earliest index on ties)
            int bi = 0; float bv = -1.f;
            for (int e = 0; e < E_NUM; ++e) {
                if (e == u0 || e == u1) continue;
                if (p[e] > bv) { bv = p[e]; bi = e; }
            }
            outg[bi] = bv;
            if (kk == 0) u0 = bi; else if (kk == 1) u1 = bi;
        }
        for (int e = 0; e < E_NUM; ++e) gm[(size_t)t * E_NUM + e] = outg[e];
    }
}

// ---------------- Kernel 2: experts (WMMA bf16) -> per-(half,eg) partials ----------------
__global__ __launch_bounds__(256) void moe_expert_kernel(
    const float* __restrict__ x,  const float* __restrict__ W1,
    const float* __restrict__ b1, const float* __restrict__ W2,
    const float* __restrict__ b2, const float* __restrict__ gm,
    float* __restrict__ partial)
{
    __shared__ __bf16 xs[TROWS * D_DIM];      // 32 KB  A for fc1, [row][d]
    __shared__ __bf16 wbuf[D_DIM * H_DIM];    // 64 KB  W1^T [h][d] or W2^T [o][h]
    __shared__ __bf16 hs[TROWS * H_DIM];      // 4  KB  A for fc2, [row][h]
    __shared__ float  gml[TROWS * 4];         // masked gates for this expert group
    __shared__ float  redbuf[2 * O_DIM];      // row-tile reduction

    const int blk  = blockIdx.x;              // 0..1023
    const int eg   = blk & 3;                 // expert group (4 experts)
    const int half = (blk >> 2) & 1;          // which 32 tokens of the batch
    const int b    = blk >> 3;                // batch index
    const int t0   = b * 64 + half * TROWS;
    const int tid  = threadIdx.x;
    const int lane = tid & 31;
    const int w    = tid >> 5;                // wave id 0..7
    const int r    = w & 1;                   // row tile (16 rows each)
    const int cq   = w >> 1;                  // 0..3 column quadrant
    const int hi8  = (lane >> 4) << 3;        // +8 rows for lanes 16..31 in C layout

    // stage x tile (bf16) and gate values
    for (int idx = tid; idx < TROWS * D_DIM; idx += 256)
        xs[idx] = f2bf(x[(size_t)t0 * D_DIM + idx]);
    for (int idx = tid; idx < TROWS * 4; idx += 256) {
        int m = idx >> 2, e = idx & 3;
        gml[idx] = gm[(size_t)(t0 + m) * E_NUM + eg * 4 + e];
    }

    v8f acc[8];
#pragma unroll
    for (int j = 0; j < 8; ++j)
#pragma unroll
        for (int i = 0; i < 8; ++i) acc[j][i] = 0.f;

    __syncthreads();

    for (int el = 0; el < 4; ++el) {
        const int e = eg * 4 + el;
        if (el < 3) {   // prefetch next expert's weights (global_prefetch_b8)
            __builtin_prefetch(&W1[((size_t)(e + 1) * D_DIM) * H_DIM + tid * 128], 0, 1);
            __builtin_prefetch(&W2[((size_t)(e + 1) * H_DIM) * O_DIM + tid * 128], 0, 1);
        }
        __syncthreads();  // prior use of wbuf complete
        // stage W1^T : [h][d], coalesced global reads
        for (int idx = tid; idx < D_DIM * H_DIM; idx += 256) {
            int d = idx >> 6, h = idx & 63;
            wbuf[h * D_DIM + d] = f2bf(W1[((size_t)e * D_DIM + d) * H_DIM + h]);
        }
        __syncthreads();

        // fc1: one 16x16 tile per wave, K = 512
        v8f c1;
#pragma unroll
        for (int i = 0; i < 8; ++i) c1[i] = 0.f;
        for (int k = 0; k < D_DIM; k += 32) {
            v16bf a  = ld_fragA(xs,   D_DIM, r * 16,  k, lane);
            v16bf bb = ld_fragB(wbuf, D_DIM, cq * 16, k, lane);
            c1 = wmma_bf16(a, bb, c1);
        }
        {
            const int hcol = cq * 16 + (lane & 15);
            const float bv = b1[(size_t)e * H_DIM + hcol];
#pragma unroll
            for (int i = 0; i < 8; ++i) {
                int m = r * 16 + i + hi8;
                float v = c1[i] + bv;
                hs[m * H_DIM + hcol] = f2bf(v > 0.f ? v : 0.f);
            }
        }
        __syncthreads();  // fc1 reads of wbuf + hs writes done

        // stage W2^T : [o][h]
        for (int idx = tid; idx < H_DIM * O_DIM; idx += 256) {
            int h = idx >> 9, o = idx & 511;
            wbuf[o * H_DIM + h] = f2bf(W2[((size_t)e * H_DIM + h) * O_DIM + o]);
        }
        __syncthreads();

        float gwv[8];
#pragma unroll
        for (int i = 0; i < 8; ++i)
            gwv[i] = gml[(r * 16 + i + hi8) * 4 + el];

        // fc2: 8 column tiles per wave, K = 64; gate-weighted accumulate (+b2)
        v16bf a0 = ld_fragA(hs, H_DIM, r * 16, 0,  lane);
        v16bf a1 = ld_fragA(hs, H_DIM, r * 16, 32, lane);
#pragma unroll
        for (int j = 0; j < 8; ++j) {
            const int n0 = (cq * 8 + j) * 16;
            v8f c2;
#pragma unroll
            for (int i = 0; i < 8; ++i) c2[i] = 0.f;
            c2 = wmma_bf16(a0, ld_fragB(wbuf, H_DIM, n0, 0,  lane), c2);
            c2 = wmma_bf16(a1, ld_fragB(wbuf, H_DIM, n0, 32, lane), c2);
            const float b2v = b2[(size_t)e * O_DIM + n0 + (lane & 15)];
#pragma unroll
            for (int i = 0; i < 8; ++i)
                acc[j][i] += gwv[i] * (c2[i] + b2v);
        }
    }

    // reduce over the 32 rows: 8 in-lane + lane L<->L+16, then combine row tiles
#pragma unroll
    for (int j = 0; j < 8; ++j) {
        float s = 0.f;
#pragma unroll
        for (int i = 0; i < 8; ++i) s += acc[j][i];
        s += __shfl_xor(s, 16);
        if (lane < 16) redbuf[r * O_DIM + (cq * 8 + j) * 16 + lane] = s;
    }
    __syncthreads();
    const size_t pbase = ((size_t)(half * 4 + eg) * 128 + b) * O_DIM;
    for (int o = tid; o < O_DIM; o += 256)
        partial[pbase + o] = redbuf[o] + redbuf[O_DIM + o];
}

// ---------------- Kernel 3: sum partials, mean over FEW ----------------
__global__ __launch_bounds__(256) void reduce_kernel(
    const float* __restrict__ partial, float* __restrict__ out)
{
    const int b = blockIdx.x;
    for (int o = threadIdx.x; o < O_DIM; o += 256) {
        float s = 0.f;
#pragma unroll
        for (int p = 0; p < 8; ++p)
            s += partial[((size_t)p * 128 + b) * O_DIM + o];
        out[(size_t)b * O_DIM + o] = s * (1.0f / 64.0f);
    }
}

extern "C" void kernel_launch(void* const* d_in, const int* in_sizes, int n_in,
                              void* d_out, int out_size, void* d_ws, size_t ws_size,
                              hipStream_t stream) {
    const float* x   = (const float*)d_in[0];
    const float* gW1 = (const float*)d_in[1];
    const float* gb1 = (const float*)d_in[2];
    const float* gW2 = (const float*)d_in[3];
    const float* gb2 = (const float*)d_in[4];
    const float* W1  = (const float*)d_in[5];
    const float* b1  = (const float*)d_in[6];
    const float* W2  = (const float*)d_in[7];
    const float* b2  = (const float*)d_in[8];

    float* gm      = (float*)d_ws;                       // [8192][16]
    float* partial = gm + (size_t)T_TOK * E_NUM;         // [8][128][512]
    float* out     = (float*)d_out;                      // [128][512]

    gate_kernel<<<T_TOK, 64, 0, stream>>>(x, gW1, gb1, gW2, gb2, gm);
    moe_expert_kernel<<<128 * 2 * 4, 256, 0, stream>>>(x, W1, b1, W2, b2, gm, partial);
    reduce_kernel<<<128, 256, 0, stream>>>(partial, out);
}
